// MoEVisionTransformer_60352880443523
// MI455X (gfx1250) — matrix operand order repaired and checked by reading, not compiled, not run on hardware
//
#include <hip/hip_runtime.h>
#include <cstdint>
#include <cstddef>

// ---------------- model constants ----------------
#define BATCH   32
#define SEQ     197
#define TOKN    (BATCH*SEQ)      // 6304
#define DM      384
#define HN      6
#define DH      64
#define MLPD    1536
#define NE      8
#define NL      6
#define NCLS    1000
#define NPATCH  196
#define PKDIM   768              // 3*16*16

typedef __bf16 bf16;
typedef __attribute__((ext_vector_type(16))) __bf16 bf16x16;
typedef __attribute__((ext_vector_type(4)))  __bf16 bf16x4;
typedef __attribute__((ext_vector_type(8)))  float  f32x8;
typedef __attribute__((ext_vector_type(4)))  float  f32x4;

__device__ inline f32x8 zero8() {
  f32x8 z;
#pragma unroll
  for (int i = 0; i < 8; ++i) z[i] = 0.f;
  return z;
}
__device__ inline f32x4 zero4() {
  f32x4 z;
#pragma unroll
  for (int i = 0; i < 4; ++i) z[i] = 0.f;
  return z;
}
__device__ inline bf16x4 zero4b() {
  bf16x4 z;
#pragma unroll
  for (int i = 0; i < 4; ++i) z[i] = (bf16)0.f;
  return z;
}
__device__ inline bf16x4 cvt4(f32x4 v) {
  bf16x4 o;
#pragma unroll
  for (int i = 0; i < 4; ++i) o[i] = (bf16)v[i];
  return o;
}

// =====================================================================
// Generic tiled WMMA GEMM:  C[M,N] = A[M,K] @ B + epilogue
//   TRANSB=true : B is [N,K] row-major (x @ W.T style)
//   TRANSB=false: B is [K,N] row-major (MoE w1/w2 style)
//   ABF16: A elements are bf16 (else f32)
//   EXPERT: grid.z selects expert; row count/gather lists per expert
//   EPI 0: C=acc+bias (f32)   1: C+=acc+bias (f32, residual)
//       2: C=bf16(gelu(w*acc+bias))   3: C=w*(acc+bias) (f32)
// Requires: K % 32 == 0, lda/ldb % 4 == 0 (true for every call site),
// so all 128-bit staging loads are in-bounds along k and 16B-aligned.
// =====================================================================
template<int EPI, bool ABF16, bool TRANSB, bool EXPERT>
__global__ __launch_bounds__(128)
void wmma_gemm(const void* __restrict__ Av, int lda,
               const float* __restrict__ Bw0, int ldb,
               const float* __restrict__ bias0,
               void* __restrict__ Cv, int ldc,
               int M, int N, int K,
               const int* __restrict__ cnts,
               const int* __restrict__ gA0,
               const int* __restrict__ gC0,
               const float* __restrict__ slotW,
               long long strideBe, int strideBiasE, int Tcap)
{
  int Mrows = M;
  const int*   gA   = gA0;
  const int*   gC   = gC0;
  const float* Bw   = Bw0;
  const float* bias = bias0;
  if (EXPERT) {
    int e = blockIdx.z;
    Mrows = cnts[e];
    gA    = gA0 + (size_t)e * Tcap;
    gC    = gC0 + (size_t)e * Tcap;
    Bw    = Bw0 + (size_t)e * strideBe;
    bias  = bias0 + (size_t)e * strideBiasE;
  }
  int m0 = blockIdx.y * 64;
  int n0 = blockIdx.x * 64;
  if (m0 >= Mrows) return;

  __shared__ bf16 As[64][32];
  __shared__ bf16 Bs[64][32];   // stored [n][k]

  int tid  = threadIdx.x;
  int lane = tid & 31;
  int wv   = tid >> 5;
  int wr   = wv >> 1, wc = wv & 1;     // wave tile origin (wr*32, wc*32)
  int half = lane >> 4, mi = lane & 15;

  f32x8 acc[2][2];
#pragma unroll
  for (int a = 0; a < 2; ++a)
#pragma unroll
    for (int b = 0; b < 2; ++b) acc[a][b] = zero8();

  for (int k0 = 0; k0 < K; k0 += 32) {
    // ---- stage A tile: 64x32, vectorized 128b loads into regs first ----
    {
      f32x4  ar[4];
      bf16x4 ab[4];
#pragma unroll
      for (int j = 0; j < 4; ++j) {
        int i = tid + j * 128;
        int r = i >> 3, c4 = (i & 7) * 4;
        int m = m0 + r;
        if (m < Mrows) {
          int ra = gA ? gA[m] : m;
          if (ABF16) ab[j] = *(const bf16x4*)((const bf16*)Av + (size_t)ra * lda + k0 + c4);
          else       ar[j] = *(const f32x4*)((const float*)Av + (size_t)ra * lda + k0 + c4);
        } else {
          if (ABF16) ab[j] = zero4b(); else ar[j] = zero4();
        }
      }
#pragma unroll
      for (int j = 0; j < 4; ++j) {
        int i = tid + j * 128;
        int r = i >> 3, c4 = (i & 7) * 4;
        *(bf16x4*)&As[r][c4] = ABF16 ? ab[j] : cvt4(ar[j]);
      }
    }
    // ---- stage B tile as [n][k] ----
    if (TRANSB) {
      f32x4 br[4];
#pragma unroll
      for (int j = 0; j < 4; ++j) {
        int i = tid + j * 128;
        int r = i >> 3, c4 = (i & 7) * 4;     // r = n, c4 = k
        br[j] = ((n0 + r) < N) ? *(const f32x4*)(Bw + (size_t)(n0 + r) * ldb + k0 + c4)
                               : zero4();
      }
#pragma unroll
      for (int j = 0; j < 4; ++j) {
        int i = tid + j * 128;
        int r = i >> 3, c4 = (i & 7) * 4;
        *(bf16x4*)&Bs[r][c4] = cvt4(br[j]);
      }
    } else {
      f32x4 br[4];
#pragma unroll
      for (int j = 0; j < 4; ++j) {
        int i = tid + j * 128;
        int kk = i >> 4, n4 = (i & 15) * 4;   // vector along n (coalesced)
        int n = n0 + n4;
        if (n + 3 < N) {
          br[j] = *(const f32x4*)(Bw + (size_t)(k0 + kk) * ldb + n);
        } else {
          f32x4 v = zero4();
#pragma unroll
          for (int jj = 0; jj < 4; ++jj)
            if (n + jj < N) v[jj] = Bw[(size_t)(k0 + kk) * ldb + n + jj];
          br[j] = v;
        }
      }
#pragma unroll
      for (int j = 0; j < 4; ++j) {
        int i = tid + j * 128;
        int kk = i >> 4, n4 = (i & 15) * 4;
#pragma unroll
        for (int jj = 0; jj < 4; ++jj) Bs[n4 + jj][kk] = (bf16)br[j][jj];
      }
    }
    __syncthreads();

    // ---- fragments per ISA layout ----
    bf16x16 af[2], bfv[2];
#pragma unroll
    for (int ms = 0; ms < 2; ++ms) {
      int row = wr * 32 + ms * 16 + mi;
#pragma unroll
      for (int i = 0; i < 8; ++i) {
        af[ms][i]     = As[row][half * 8 + i];
        af[ms][i + 8] = As[row][16 + half * 8 + i];
      }
    }
#pragma unroll
    for (int ns = 0; ns < 2; ++ns) {
      int row = wc * 32 + ns * 16 + mi;
#pragma unroll
      for (int i = 0; i < 8; ++i) {
        bfv[ns][i]     = Bs[row][half * 8 + i];
        bfv[ns][i + 8] = Bs[row][16 + half * 8 + i];
      }
    }
#pragma unroll
    for (int ms = 0; ms < 2; ++ms)
#pragma unroll
      for (int ns = 0; ns < 2; ++ns)
        acc[ms][ns] = __builtin_amdgcn_wmma_f32_16x16x32_bf16(
            false, af[ms], false, bfv[ns], (short)0, acc[ms][ns], false, false);
    __syncthreads();
  }

  // ---- epilogue ----
#pragma unroll
  for (int ms = 0; ms < 2; ++ms) {
#pragma unroll
    for (int ns = 0; ns < 2; ++ns) {
#pragma unroll
      for (int r = 0; r < 8; ++r) {
        int m = m0 + wr * 32 + ms * 16 + r + 8 * half;
        int n = n0 + wc * 32 + ns * 16 + mi;
        if (m < Mrows && n < N) {
          float a  = acc[ms][ns][r];
          int   mc = gC ? gC[m] : m;
          float bv = bias ? bias[n] : 0.f;
          if (EPI == 0) {
            ((float*)Cv)[(size_t)mc * ldc + n] = a + bv;
          } else if (EPI == 1) {
            ((float*)Cv)[(size_t)mc * ldc + n] += a + bv;
          } else if (EPI == 2) {
            float w = slotW[mc];
            float t = w * a + bv;
            float g = 0.5f * t * (1.f + erff(t * 0.70710678118654752f));
            ((bf16*)Cv)[(size_t)mc * ldc + n] = (bf16)g;
          } else { // EPI == 3
            float w = slotW[mc];
            ((float*)Cv)[(size_t)mc * ldc + n] = w * (a + bv);
          }
        }
      }
    }
  }
}

// =====================================================================
// Attention: one wave per (q-tile, head, batch). qkv is [T,1152] f32.
// Writes ctx[T, 384] (head-concatenated context).
// =====================================================================
__global__ __launch_bounds__(32)
void attn_kernel(const float* __restrict__ qkv, float* __restrict__ ctx)
{
  int qt = blockIdx.x;        // 0..12
  int h  = blockIdx.y;        // 0..5
  int b  = blockIdx.z;        // 0..31
  int tokbase = b * SEQ;

  __shared__ bf16 Qs[16][64];
  __shared__ bf16 Ks[16][64];
  __shared__ bf16 Pn[16][224];   // probs, padded to 224 cols
  __shared__ bf16 Vn[64][32];    // V tile transposed [d][s]

  int tid  = threadIdx.x;
  int half = tid >> 4, mi = tid & 15;

  // Q tile, pre-scaled by dh^-0.5 = 0.125 (exact in bf16); 128b staged loads
  {
    f32x4 qr[8];
#pragma unroll
    for (int j = 0; j < 8; ++j) {
      int i = tid + j * 32;
      int r = i >> 4, d4 = (i & 15) * 4;
      int q = qt * 16 + r;
      qr[j] = (q < SEQ)
            ? *(const f32x4*)(qkv + (size_t)(tokbase + q) * 1152 + h * 64 + d4)
            : zero4();
    }
#pragma unroll
    for (int j = 0; j < 8; ++j) {
      int i = tid + j * 32;
      int r = i >> 4, d4 = (i & 15) * 4;
      *(bf16x4*)&Qs[r][d4] = cvt4(qr[j] * 0.125f);
    }
  }
  __syncthreads();

  bf16x16 aq[2];
#pragma unroll
  for (int ks = 0; ks < 2; ++ks)
#pragma unroll
    for (int i = 0; i < 8; ++i) {
      aq[ks][i]     = Qs[mi][ks * 32 + half * 8 + i];
      aq[ks][i + 8] = Qs[mi][ks * 32 + 16 + half * 8 + i];
    }

  // scores: 13 k-tiles of 16, kept in registers
  f32x8 sc[13];
#pragma unroll
  for (int kt = 0; kt < 13; ++kt) {
    __syncthreads();
    {
      f32x4 kr[8];
#pragma unroll
      for (int j = 0; j < 8; ++j) {
        int i = tid + j * 32;
        int r = i >> 4, d4 = (i & 15) * 4;
        int s = kt * 16 + r;
        kr[j] = (s < SEQ)
              ? *(const f32x4*)(qkv + (size_t)(tokbase + s) * 1152 + 384 + h * 64 + d4)
              : zero4();
      }
#pragma unroll
      for (int j = 0; j < 8; ++j) {
        int i = tid + j * 32;
        int r = i >> 4, d4 = (i & 15) * 4;
        *(bf16x4*)&Ks[r][d4] = cvt4(kr[j]);
      }
    }
    __syncthreads();
    f32x8 c = zero8();
#pragma unroll
    for (int ks = 0; ks < 2; ++ks) {
      bf16x16 bk;
#pragma unroll
      for (int i = 0; i < 8; ++i) {
        bk[i]     = Ks[mi][ks * 32 + half * 8 + i];
        bk[i + 8] = Ks[mi][ks * 32 + 16 + half * 8 + i];
      }
      c = __builtin_amdgcn_wmma_f32_16x16x32_bf16(false, aq[ks], false, bk,
                                                  (short)0, c, false, false);
    }
    sc[kt] = c;
  }

  // masked softmax, row-wise (row = r + 8*half, cols spread over 16 lanes of half)
  float invr[8];
#pragma unroll
  for (int r = 0; r < 8; ++r) {
    float mx = -1e30f;
#pragma unroll
    for (int kt = 0; kt < 13; ++kt) {
      int n = kt * 16 + mi;
      float v = (n < SEQ) ? sc[kt][r] : -1e30f;
      mx = fmaxf(mx, v);
    }
#pragma unroll
    for (int m2 = 8; m2 >= 1; m2 >>= 1) mx = fmaxf(mx, __shfl_xor(mx, m2, 32));
    float sum = 0.f;
#pragma unroll
    for (int kt = 0; kt < 13; ++kt) {
      int n = kt * 16 + mi;
      float p = (n < SEQ) ? __expf(sc[kt][r] - mx) : 0.f;
      sc[kt][r] = p;
      sum += p;
    }
#pragma unroll
    for (int m2 = 8; m2 >= 1; m2 >>= 1) sum += __shfl_xor(sum, m2, 32);
    invr[r] = 1.f / sum;
  }

  // write P to LDS (zero padding cols first, 64b stores)
  {
    bf16x4 z = zero4b();
    for (int i = tid; i < 16 * 224 / 4; i += 32) ((bf16x4*)&Pn[0][0])[i] = z;
  }
  __syncthreads();
#pragma unroll
  for (int kt = 0; kt < 13; ++kt)
#pragma unroll
    for (int r = 0; r < 8; ++r)
      Pn[r + 8 * half][kt * 16 + mi] = (bf16)(sc[kt][r] * invr[r]);
  __syncthreads();

  // O = P[16,224] @ V[224,64]
  f32x8 o[4];
#pragma unroll
  for (int nt = 0; nt < 4; ++nt) o[nt] = zero8();

  for (int ks2 = 0; ks2 < 7; ++ks2) {
    __syncthreads();
    {
      f32x4 vr[16];
#pragma unroll
      for (int j = 0; j < 16; ++j) {
        int i = tid + j * 32;
        int ss = i >> 4, d4 = (i & 15) * 4;
        int s = ks2 * 32 + ss;
        vr[j] = (s < SEQ)
              ? *(const f32x4*)(qkv + (size_t)(tokbase + s) * 1152 + 768 + h * 64 + d4)
              : zero4();
      }
#pragma unroll
      for (int j = 0; j < 16; ++j) {
        int i = tid + j * 32;
        int ss = i >> 4, d4 = (i & 15) * 4;
#pragma unroll
        for (int jj = 0; jj < 4; ++jj) Vn[d4 + jj][ss] = (bf16)vr[j][jj];
      }
    }
    __syncthreads();
    bf16x16 ap;
#pragma unroll
    for (int i = 0; i < 8; ++i) {
      ap[i]     = Pn[mi][ks2 * 32 + half * 8 + i];
      ap[i + 8] = Pn[mi][ks2 * 32 + 16 + half * 8 + i];
    }
#pragma unroll
    for (int nt = 0; nt < 4; ++nt) {
      bf16x16 bv;
#pragma unroll
      for (int i = 0; i < 8; ++i) {
        bv[i]     = Vn[nt * 16 + mi][half * 8 + i];
        bv[i + 8] = Vn[nt * 16 + mi][16 + half * 8 + i];
      }
      o[nt] = __builtin_amdgcn_wmma_f32_16x16x32_bf16(false, ap, false, bv,
                                                      (short)0, o[nt], false, false);
    }
  }

#pragma unroll
  for (int nt = 0; nt < 4; ++nt)
#pragma unroll
    for (int r = 0; r < 8; ++r) {
      int q = qt * 16 + r + 8 * half;
      if (q < SEQ)
        ctx[(size_t)(tokbase + q) * DM + h * 64 + nt * 16 + mi] = o[nt][r];
    }
}

// =====================================================================
// LayerNorm (row-strided so it also handles the final cls-only LN)
// =====================================================================
__global__ __launch_bounds__(128)
void ln_kernel(const float* __restrict__ x, float* __restrict__ y,
               const float* __restrict__ g, const float* __restrict__ b,
               int rows, long long xstride, long long ystride)
{
  int row = blockIdx.x;
  if (row >= rows) return;
  const float* xr = x + (size_t)row * xstride;
  float*       yr = y + (size_t)row * ystride;
  __shared__ float red[128];
  int tid = threadIdx.x;

  float s = 0.f;
  for (int i = tid; i < DM; i += 128) s += xr[i];
  red[tid] = s; __syncthreads();
  for (int o = 64; o > 0; o >>= 1) { if (tid < o) red[tid] += red[tid + o]; __syncthreads(); }
  float mean = red[0] * (1.f / DM);
  __syncthreads();

  float v = 0.f;
  for (int i = tid; i < DM; i += 128) { float d = xr[i] - mean; v += d * d; }
  red[tid] = v; __syncthreads();
  for (int o = 64; o > 0; o >>= 1) { if (tid < o) red[tid] += red[tid + o]; __syncthreads(); }
  float rstd = rsqrtf(red[0] * (1.f / DM) + 1e-5f);
  __syncthreads();

  for (int i = tid; i < DM; i += 128)
    yr[i] = (xr[i] - mean) * rstd * g[i] + b[i];
}

// =====================================================================
// small elementwise / dispatch kernels
// =====================================================================
__global__ void im2col_kernel(const float* __restrict__ x, float* __restrict__ patches)
{
  int idx = blockIdx.x * blockDim.x + threadIdx.x;
  if (idx >= BATCH * NPATCH * PKDIM) return;
  int m = idx / PKDIM, k = idx % PKDIM;
  int b = m / NPATCH, pp = m % NPATCH;
  int hp = pp / 14, wp = pp % 14;
  int c = k / 256, r2 = k % 256, i = r2 / 16, j = r2 % 16;
  patches[idx] = x[(((size_t)(b * 3 + c) * 224) + (hp * 16 + i)) * 224 + wp * 16 + j];
}

__global__ void assemble_kernel(const float* __restrict__ ptmp,
                                const float* __restrict__ cls,
                                const float* __restrict__ pos,
                                float* __restrict__ tok)
{
  int idx = blockIdx.x * blockDim.x + threadIdx.x;
  if (idx >= TOKN * DM) return;
  int t = idx / DM, d = idx % DM;
  int b = t / SEQ, s = t % SEQ;
  float v = (s == 0) ? cls[d] : ptmp[(size_t)(b * NPATCH + s - 1) * DM + d];
  tok[idx] = v + pos[(size_t)s * DM + d];
}

__global__ void gate_kernel(const float* __restrict__ xn,
                            const float* __restrict__ gw,
                            const float* __restrict__ gb,
                            float* __restrict__ gl)
{
  int idx = blockIdx.x * blockDim.x + threadIdx.x;
  if (idx >= TOKN * NE) return;
  int t = idx >> 3, e = idx & 7;
  const f32x4* xr = (const f32x4*)(xn + (size_t)t * DM);
  const f32x4* wr = (const f32x4*)(gw + (size_t)e * DM);
  float acc = gb[e];
  for (int k = 0; k < DM / 4; ++k) {
    f32x4 a = xr[k], w = wr[k];
#pragma unroll
    for (int j = 0; j < 4; ++j) acc += a[j] * w[j];
  }
  gl[idx] = acc;
}

__global__ void zero_counts_kernel(int* cnt)
{
  if (threadIdx.x < NE) cnt[threadIdx.x] = 0;
}

__global__ void dispatch_kernel(const float* __restrict__ gl,
                                int* __restrict__ cnt,
                                int* __restrict__ listA,
                                int* __restrict__ listC,
                                float* __restrict__ slotW)
{
  int t = blockIdx.x * blockDim.x + threadIdx.x;
  if (t >= TOKN) return;
  float v[NE];
#pragma unroll
  for (int e = 0; e < NE; ++e) v[e] = gl[t * NE + e];
  int i1 = 0; float b1 = v[0];
#pragma unroll
  for (int e = 1; e < NE; ++e) if (v[e] > b1) { b1 = v[e]; i1 = e; }
  int i2 = -1; float b2 = -1e30f;
#pragma unroll
  for (int e = 0; e < NE; ++e) if (e != i1 && v[e] > b2) { b2 = v[e]; i2 = e; }
  float e2 = __expf(b2 - b1);
  float w1 = 1.f / (1.f + e2);
  float w2 = e2 * w1;
  slotW[2 * t]     = w1;
  slotW[2 * t + 1] = w2;
  int p = atomicAdd(&cnt[i1], 1);
  listA[(size_t)i1 * TOKN + p] = t;
  listC[(size_t)i1 * TOKN + p] = 2 * t;
  p = atomicAdd(&cnt[i2], 1);
  listA[(size_t)i2 * TOKN + p] = t;
  listC[(size_t)i2 * TOKN + p] = 2 * t + 1;
}

__global__ void combine_kernel(const float* __restrict__ Ob, float* __restrict__ tok)
{
  int idx = blockIdx.x * blockDim.x + threadIdx.x;
  if (idx >= TOKN * DM) return;
  int t = idx / DM, n = idx % DM;
  tok[idx] += Ob[(size_t)(2 * t) * DM + n] + Ob[(size_t)(2 * t + 1) * DM + n];
}

// =====================================================================
// host orchestration
// =====================================================================
extern "C" void kernel_launch(void* const* d_in, const int* in_sizes, int n_in,
                              void* d_out, int out_size, void* d_ws, size_t ws_size,
                              hipStream_t stream)
{
  (void)in_sizes; (void)n_in; (void)out_size; (void)ws_size;

  const float* x        = (const float*)d_in[0];
  const float* patch_w  = (const float*)d_in[1];
  const float* patch_b  = (const float*)d_in[2];
  const float* cls_tok  = (const float*)d_in[3];
  const float* pos_emb  = (const float*)d_in[4];
  const float* ln1_g    = (const float*)d_in[5];
  const float* ln1_b    = (const float*)d_in[6];
  const float* qkv_w    = (const float*)d_in[7];
  const float* qkv_b    = (const float*)d_in[8];
  const float* out_w    = (const float*)d_in[9];
  const float* out_b    = (const float*)d_in[10];
  const float* ln2_g    = (const float*)d_in[11];
  const float* ln2_b    = (const float*)d_in[12];
  const float* gate_w   = (const float*)d_in[13];
  const float* gate_b   = (const float*)d_in[14];
  const float* w1       = (const float*)d_in[15];
  const float* b1       = (const float*)d_in[16];
  const float* w2       = (const float*)d_in[17];
  const float* b2       = (const float*)d_in[18];
  const float* lnf_g    = (const float*)d_in[19];
  const float* lnf_b    = (const float*)d_in[20];
  const float* head_w   = (const float*)d_in[21];
  const float* head_b   = (const float*)d_in[22];

  float* out = (float*)d_out;          // [32*1000 logits][L*T*E gates]

  // -------- workspace carve --------
  char* w = (char*)d_ws;
  size_t oTok  = 0;
  size_t oXn   = oTok  + (size_t)TOKN * DM * 4;
  size_t oQkv  = oXn   + (size_t)TOKN * DM * 4;
  size_t oCtx  = oQkv  + (size_t)TOKN * 1152 * 4;
  size_t oObuf = oCtx  + (size_t)TOKN * DM * 4;
  size_t oH    = oObuf + (size_t)TOKN * 2 * DM * 4;
  size_t oSlW  = oH    + (size_t)TOKN * 2 * MLPD * 2;
  size_t oLA   = oSlW  + (size_t)TOKN * 2 * 4;
  size_t oLC   = oLA   + (size_t)NE * TOKN * 4;
  size_t oCnt  = oLC   + (size_t)NE * TOKN * 4;

  float* tok   = (float*)(w + oTok);
  float* xn    = (float*)(w + oXn);
  float* qkvb  = (float*)(w + oQkv);
  float* ctx   = (float*)(w + oCtx);
  float* Obuf  = (float*)(w + oObuf);
  bf16*  Hb    = (bf16*)(w + oH);
  float* slotW = (float*)(w + oSlW);
  int*   listA = (int*)(w + oLA);
  int*   listC = (int*)(w + oLC);
  int*   cnt   = (int*)(w + oCnt);
  float* patches = qkvb;   // alias: used only before first qkv GEMM
  float* ptmp    = Obuf;   // alias: patch embed output before layer loop
  float* xf      = ctx;    // alias: final LN output (after last attention use)

  const int rowT = (TOKN + 63) / 64;   // 99

  // ---- patch embedding ----
  im2col_kernel<<<(BATCH * NPATCH * PKDIM + 255) / 256, 256, 0, stream>>>(x, patches);
  wmma_gemm<0, false, true, false><<<dim3(DM / 64, (BATCH * NPATCH + 63) / 64, 1), 128, 0, stream>>>(
      patches, PKDIM, patch_w, PKDIM, patch_b, ptmp, DM,
      BATCH * NPATCH, DM, PKDIM, nullptr, nullptr, nullptr, nullptr, 0, 0, 0);
  assemble_kernel<<<(TOKN * DM + 255) / 256, 256, 0, stream>>>(ptmp, cls_tok, pos_emb, tok);

  // ---- transformer layers ----
  for (int l = 0; l < NL; ++l) {
    const float* qw  = qkv_w  + (size_t)l * 3 * DM * DM;
    const float* qb  = qkv_b  + (size_t)l * 3 * DM;
    const float* ow  = out_w  + (size_t)l * DM * DM;
    const float* ob  = out_b  + (size_t)l * DM;
    const float* gw  = gate_w + (size_t)l * NE * DM;
    const float* gb  = gate_b + (size_t)l * NE;
    const float* w1l = w1     + (size_t)l * NE * DM * MLPD;
    const float* b1l = b1     + (size_t)l * NE * MLPD;
    const float* w2l = w2     + (size_t)l * NE * MLPD * DM;
    const float* b2l = b2     + (size_t)l * NE * DM;
    float* glout = out + 32 * NCLS + (size_t)l * TOKN * NE;

    // MHA
    ln_kernel<<<TOKN, 128, 0, stream>>>(tok, xn, ln1_g + l * DM, ln1_b + l * DM, TOKN, DM, DM);
    wmma_gemm<0, false, true, false><<<dim3(3 * DM / 64, rowT, 1), 128, 0, stream>>>(
        xn, DM, qw, DM, qb, qkvb, 3 * DM, TOKN, 3 * DM, DM,
        nullptr, nullptr, nullptr, nullptr, 0, 0, 0);
    attn_kernel<<<dim3(13, HN, BATCH), 32, 0, stream>>>(qkvb, ctx);
    wmma_gemm<1, false, true, false><<<dim3(DM / 64, rowT, 1), 128, 0, stream>>>(
        ctx, DM, ow, DM, ob, tok, DM, TOKN, DM, DM,
        nullptr, nullptr, nullptr, nullptr, 0, 0, 0);

    // MoE
    ln_kernel<<<TOKN, 128, 0, stream>>>(tok, xn, ln2_g + l * DM, ln2_b + l * DM, TOKN, DM, DM);
    gate_kernel<<<(TOKN * NE + 255) / 256, 256, 0, stream>>>(xn, gw, gb, glout);
    zero_counts_kernel<<<1, 64, 0, stream>>>(cnt);
    dispatch_kernel<<<(TOKN + 255) / 256, 256, 0, stream>>>(glout, cnt, listA, listC, slotW);
    // up-proj + gelu -> H (bf16):  h = gelu(w * (x @ W1_e) + b1_e)
    wmma_gemm<2, false, false, true><<<dim3(MLPD / 64, rowT, NE), 128, 0, stream>>>(
        xn, DM, w1l, MLPD, b1l, Hb, MLPD, TOKN, MLPD, DM,
        cnt, listA, listC, slotW, (long long)DM * MLPD, MLPD, TOKN);
    // down-proj:  o = w * (h @ W2_e + b2_e)
    wmma_gemm<3, true, false, true><<<dim3(DM / 64, rowT, NE), 128, 0, stream>>>(
        Hb, MLPD, w2l, DM, b2l, Obuf, DM, TOKN, DM, MLPD,
        cnt, listC, listC, slotW, (long long)MLPD * DM, DM, TOKN);
    combine_kernel<<<(TOKN * DM + 255) / 256, 256, 0, stream>>>(Obuf, tok);
  }

  // ---- head (cls token only) ----
  ln_kernel<<<BATCH, 128, 0, stream>>>(tok, xf, lnf_g, lnf_b, BATCH, (long long)SEQ * DM, DM);
  wmma_gemm<0, false, true, false><<<dim3((NCLS + 63) / 64, 1, 1), 128, 0, stream>>>(
      xf, DM, head_w, DM, head_b, out, NCLS, BATCH, NCLS, DM,
      nullptr, nullptr, nullptr, nullptr, 0, 0, 0);
}
